// FKAConvNetwork_69226282877016
// MI455X (gfx1250) — matrix operand order repaired and checked by reading, not compiled, not run on hardware
//
#include <hip/hip_runtime.h>
#include <hip/hip_bf16.h>

// ---------------------------------------------------------------------------
// FKAConv network for MI455X (gfx1250, wave32).
// All channel-mixing contractions run on the f16 WMMA pipe
// (v_wmma_f32_16x16x32_f16, fp32 accumulate) with pre-packed zero-padded f16
// operands so the hot loop is pure b128 loads + WMMA (no guards, no cvt).
// Irregular ops (FPS, KNN, gathers, instance-norm stats) run on VALU/LDS.
// ---------------------------------------------------------------------------

typedef __attribute__((ext_vector_type(16))) _Float16 v16h;
typedef __attribute__((ext_vector_type(8)))  _Float16 v8h;
typedef __attribute__((ext_vector_type(8)))  float    v8f;

#define BN_INV 0.99999500003749968751f   // 1/sqrt(1+1e-5)
#define BATCH 2

// ---------------------------------------------------------------------------
// WMMA GEMM on packed f16 operands.
//   Wh: [Mr, Kp] f16 row-major (zero-padded K)
//   Xh: [Nc, Kp] f16 point-major (zero-padded K)
//   Y : activation layout [B, Mr, Nper] f32  (col j -> b=j/Nper, n=j%Nper)
// Wave computes a 32x16 tile (two 16x16 WMMA accumulators sharing B).
// Requires: Mr % 32 == 0, Nc % 64 == 0, Kp % 32 == 0 (guaranteed by packing).
// ---------------------------------------------------------------------------
__global__ __launch_bounds__(128)
void k_gemm_wmma(const _Float16* __restrict__ Wh, const _Float16* __restrict__ Xh,
                 float* __restrict__ Y,
                 const float* __restrict__ bias, const float* __restrict__ bng,
                 const float* __restrict__ bnb,
                 int Mr, int Kp, int Nc, int Nper, int relu) {
  int lane  = threadIdx.x & 31;
  int wave  = threadIdx.x >> 5;
  int ntile = blockIdx.x * 4 + wave;
  int mtile = blockIdx.y;                       // 32 output rows per tile
  if (ntile * 16 >= Nc) return;                 // wave-uniform (never taken)

  int half = lane >> 4;
  int lm   = lane & 15;

  const _Float16* A0 = Wh + ((long)(mtile * 32) + lm) * Kp;   // rows m..m+15
  const _Float16* A1 = A0 + (long)16 * Kp;                    // rows m+16..m+31
  const _Float16* Bc = Xh + ((long)(ntile * 16) + lm) * Kp;   // column

  v8f acc0 = {0.f, 0.f, 0.f, 0.f, 0.f, 0.f, 0.f, 0.f};
  v8f acc1 = {0.f, 0.f, 0.f, 0.f, 0.f, 0.f, 0.f, 0.f};

  for (int kk = 0; kk < Kp; kk += 32) {
    __builtin_prefetch(A0 + kk + 256, 0, 3);
    __builtin_prefetch(Bc + kk + 256, 0, 3);
    // A fragment: halves K in [kk+8h, kk+8h+8) and [kk+16+8h, kk+16+8h+8)
    v8h a0l = *(const v8h*)(A0 + kk + 8 * half);
    v8h a0h = *(const v8h*)(A0 + kk + 16 + 8 * half);
    v8h a1l = *(const v8h*)(A1 + kk + 8 * half);
    v8h a1h = *(const v8h*)(A1 + kk + 16 + 8 * half);
    // B fragment: halves K in [kk+8h, kk+8h+8) and [kk+16+8h, kk+16+8h+8)
    v8h b0  = *(const v8h*)(Bc + kk + 8 * half);
    v8h b1  = *(const v8h*)(Bc + kk + 16 + 8 * half);
    v16h af0 = __builtin_shufflevector(a0l, a0h, 0, 1, 2, 3, 4, 5, 6, 7,
                                       8, 9, 10, 11, 12, 13, 14, 15);
    v16h af1 = __builtin_shufflevector(a1l, a1h, 0, 1, 2, 3, 4, 5, 6, 7,
                                       8, 9, 10, 11, 12, 13, 14, 15);
    v16h bf  = __builtin_shufflevector(b0, b1, 0, 1, 2, 3, 4, 5, 6, 7,
                                       8, 9, 10, 11, 12, 13, 14, 15);
    acc0 = __builtin_amdgcn_wmma_f32_16x16x32_f16(
        false, af0, false, bf, (short)0, acc0, false, false);
    acc1 = __builtin_amdgcn_wmma_f32_16x16x32_f16(
        false, af1, false, bf, (short)0, acc1, false, false);
  }

  int bcol = ntile * 16 + lm;
  int bb = bcol / Nper, nn = bcol % Nper;
  float* Yc = Y + (long)bb * Mr * Nper + nn;
#pragma unroll
  for (int r = 0; r < 8; ++r) {
#pragma unroll
    for (int t2 = 0; t2 < 2; ++t2) {
      int orow = mtile * 32 + t2 * 16 + r + half * 8;
      float s = 1.f, sh = 0.f;
      if (bng)  s  = bng[orow] * BN_INV;
      if (bias) sh = bias[orow] * s;
      if (bnb)  sh += bnb[orow];
      float v = (t2 ? acc1[r] : acc0[r]) * s + sh;
      if (relu) v = fmaxf(v, 0.f);
      Yc[(long)orow * Nper] = v;
    }
  }
}

// ---------------------------------------------------------------------------
// Operand packing: f32 -> zero-padded f16
// ---------------------------------------------------------------------------
__global__ void k_pack_w(const float* __restrict__ W, _Float16* __restrict__ Wh,
                         int Kd, int Kp, int total) {
  int g = blockIdx.x * 256 + threadIdx.x;       // over Mr*Kp
  if (g >= total) return;
  int k = g % Kp, r = g / Kp;
  Wh[g] = (k < Kd) ? (_Float16)W[(long)r * Kd + k] : (_Float16)0.f;
}

__global__ void k_pack_x(const float* __restrict__ X, _Float16* __restrict__ Xh,
                         int Kd, int Kp, int Nper, int total) {
  int g = blockIdx.x * 256 + threadIdx.x;       // over Nc*Kp
  if (g >= total) return;
  int k = g % Kp, j = g / Kp;
  int b = j / Nper, n = j % Nper;
  Xh[g] = (k < Kd) ? (_Float16)X[((long)b * Kd + k) * Nper + n] : (_Float16)0.f;
}

// ---------------------------------------------------------------------------
// Elementwise / data-movement kernels
// ---------------------------------------------------------------------------
__global__ void k_zero(float* p, int n) {
  int g = blockIdx.x * 256 + threadIdx.x;
  if (g < n) p[g] = 0.f;
}

__global__ void k_transpose(const float* in, float* out, int N, int C) {
  int g = blockIdx.x * 256 + threadIdx.x;           // over BATCH*N*C
  if (g < BATCH * N * C) {
    int c = g % C, n = (g / C) % N, b = g / (N * C);
    out[((long)b * C + c) * N + n] = in[g];
  }
}

__global__ void k_gather3(const float* pos, const int* idx, int Np, int Ns, float* out) {
  int g = blockIdx.x * 256 + threadIdx.x;           // over BATCH*3*Ns
  if (g < BATCH * 3 * Ns) {
    int n = g % Ns, d = (g / Ns) % 3, b = g / (3 * Ns);
    out[g] = pos[((long)b * 3 + d) * Np + idx[b * Ns + n]];
  }
}

__global__ void k_maxpool(const float* x, const int* idx, int C, int Np, int Ns, float* y) {
  int g = blockIdx.x * 256 + threadIdx.x;           // over BATCH*C*Ns
  if (g >= BATCH * C * Ns) return;
  int n = g % Ns, c = (g / Ns) % C, b = g / (Ns * C);
  const int* id = idx + ((long)b * Ns + n) * 16;
  const float* xr = x + ((long)b * C + c) * Np;
  float m = -3.4e38f;
#pragma unroll
  for (int j = 0; j < 16; ++j) m = fmaxf(m, xr[id[j]]);
  y[((long)b * C + c) * Ns + n] = m;
}

__global__ void k_add_relu(const float* a, const float* b, float* y, int n) {
  int g = blockIdx.x * 256 + threadIdx.x;
  if (g < n) y[g] = fmaxf(a[g] + b[g], 0.f);
}

__global__ void k_mean(const float* x, int C, int Ns, float* y) {
  int g = blockIdx.x * 256 + threadIdx.x;           // over BATCH*C
  if (g < BATCH * C) {
    int b = g / C, c = g % C;
    const float* xr = x + ((long)b * C + c) * Ns;
    float s = 0.f;
    for (int n = 0; n < Ns; ++n) s += xr[n];
    y[g] = s / (float)Ns;
  }
}

__global__ void k_fc(const float* xm, const float* w, const float* bias,
                     float* out, int C, int NC) {
  int g = blockIdx.x * 64 + threadIdx.x;            // over BATCH*NC
  if (g >= BATCH * NC) return;
  int b = g / NC, c = g % NC;
  const float* xr = xm + (long)b * C;
  const float* wr = w + (long)c * C;
  float s = bias[c];
  for (int i = 0; i < C; ++i) s += xr[i] * wr[i];
  out[g] = s;
}

// ---------------------------------------------------------------------------
// Farthest point sampling: one block per batch, sequential argmax iterations.
// ---------------------------------------------------------------------------
__global__ void k_fps(const float* pos, int Np, int n, int* idxs, float* mind) {
  int b = blockIdx.x;
  const float* P = pos + (long)b * 3 * Np;
  int* ids = idxs + b * n;
  float* md = mind + (long)b * Np;
  int t = threadIdx.x;
  __shared__ float sv[256];
  __shared__ int   si[256];
  for (int j = t; j < Np; j += 256) md[j] = 1e10f;
  if (t == 0) ids[0] = 0;
  __syncthreads();
  int last = 0;
  for (int i = 1; i < n; ++i) {
    float lx = P[last], ly = P[Np + last], lz = P[2 * Np + last];
    float bv = -1.f; int bi = 0;
    for (int j = t; j < Np; j += 256) {
      float dx = P[j] - lx, dy = P[Np + j] - ly, dz = P[2 * Np + j] - lz;
      float d = dx * dx + dy * dy + dz * dz;
      float m = fminf(md[j], d);
      md[j] = m;
      if (m > bv) { bv = m; bi = j; }
    }
    sv[t] = bv; si[t] = bi;
    __syncthreads();
    for (int s = 128; s > 0; s >>= 1) {
      if (t < s) {
        if (sv[t + s] > sv[t] || (sv[t + s] == sv[t] && si[t + s] < si[t])) {
          sv[t] = sv[t + s]; si[t] = si[t + s];
        }
      }
      __syncthreads();
    }
    last = si[0];
    if (t == 0) ids[i] = last;
    __syncthreads();
  }
}

// ---------------------------------------------------------------------------
// Brute-force KNN, one thread per query, sorted top-16 (nearest first).
// ---------------------------------------------------------------------------
__global__ void k_knn(const float* pts, const float* qs, int Np, int Nq, int* out) {
  int g = blockIdx.x * 256 + threadIdx.x;
  if (g >= BATCH * Nq) return;
  int b = g / Nq, q = g % Nq;
  const float* P = pts + (long)b * 3 * Np;
  float qx = qs[(long)b * 3 * Nq + q];
  float qy = qs[(long)b * 3 * Nq + Nq + q];
  float qz = qs[(long)b * 3 * Nq + 2 * Nq + q];
  float bd[16]; int bi_[16];
#pragma unroll
  for (int m = 0; m < 16; ++m) { bd[m] = 3.4e38f; bi_[m] = 0; }
  for (int j = 0; j < Np; ++j) {
    float dx = P[j] - qx, dy = P[Np + j] - qy, dz = P[2 * Np + j] - qz;
    float d = dx * dx + dy * dy + dz * dz;
    if (d < bd[15]) {
      int m = 15;
      while (m > 0 && bd[m - 1] > d) { bd[m] = bd[m - 1]; bi_[m] = bi_[m - 1]; --m; }
      bd[m] = d; bi_[m] = j;
    }
  }
#pragma unroll
  for (int m = 0; m < 16; ++m) out[((long)b * Nq + q) * 16 + m] = bi_[m];
}

// ---------------------------------------------------------------------------
// FKAConv stage A: local coords, dist, adaptive weights dw, y1 = fc1*pts,
// plus instance-norm partial sums for y1. Block = 16 points x 16 neighbors.
// mat layout: point-major [pt][k][m].
// ---------------------------------------------------------------------------
__global__ void k_fka_a(const float* pos, const float* sup, const int* idx,
                        const float* fc1, const float* alpha, const float* beta,
                        int Np, int Ns, float* dw, float* mat,
                        float* st1, float* st2) {
  __shared__ float sdw[16][16];
  __shared__ float sSum[16], sSq[16];
  int t = threadIdx.x, pl = t >> 4, m = t & 15;
  int P = blockIdx.x * 16 + pl;
  int b = P / Ns, n = P % Ns;
  if (t < 16) { sSum[t] = 0.f; sSq[t] = 0.f; }
  int i = idx[(long)P * 16 + m];
  float sx = sup[((long)b * 3 + 0) * Ns + n];
  float sy = sup[((long)b * 3 + 1) * Ns + n];
  float sz = sup[((long)b * 3 + 2) * Ns + n];
  float px = pos[((long)b * 3 + 0) * Np + i] - sx;
  float py = pos[((long)b * 3 + 1) * Np + i] - sy;
  float pz = pos[((long)b * 3 + 2) * Np + i] - sz;
  float dist = sqrtf(px * px + py * py + pz * pz);
  float dr = 1.f / (1.f + __expf(alpha[0] * dist - beta[0]));
  sdw[pl][m] = dr;
  __syncthreads();
  float dws = 0.f;
  for (int mm = 0; mm < 16; ++mm) dws += sdw[pl][mm];
  dws += (dws == 0.f ? 1.f : 0.f) + 1e-6f;
  float dwv = dr / dws * 16.f;
  dw[(long)P * 16 + m] = dwv;
#pragma unroll
  for (int k = 0; k < 16; ++k) {
    float y = fc1[k * 3 + 0] * px + fc1[k * 3 + 1] * py + fc1[k * 3 + 2] * pz;
    mat[(long)P * 256 + k * 16 + m] = y;
    atomicAdd(&sSum[k], y);
    atomicAdd(&sSq[k], y * y);
  }
  __syncthreads();
  int b0 = (blockIdx.x * 16) / Ns;
  if (t < 16) {
    atomicAdd(&st1[b0 * 16 + t], sSum[t]);
    atomicAdd(&st2[b0 * 16 + t], sSq[t]);
  }
}

// finalize instance-norm stats; zero accumulators for the next stage
__global__ void k_stats(float* st1, float* st2, int count, float* mean, float* rstd) {
  int t = threadIdx.x;
  if (t < 32) {
    float mu = st1[t] / (float)count;
    float v  = st2[t] / (float)count - mu * mu;
    mean[t] = mu;
    rstd[t] = rsqrtf(fmaxf(v, 0.f) + 1e-5f);
    st1[t] = 0.f; st2[t] = 0.f;
  }
}

// ---------------------------------------------------------------------------
// FKAConv stages B/C (in place on mat):
//   normalize(prev stats) -> relu -> mp = max_m(mat*dw) -> cat(mat, mp)
//   -> y = fc * cat    [stage B: store raw + stats; stage C: relu * dw]
// Block = 16 points x 16 output channels.
// ---------------------------------------------------------------------------
__global__ void k_fka_bc(const float* mean, const float* rstd,
                         const float* g, const float* bb, const float* fc,
                         const float* dw, float* mat, int Ns,
                         float* st1, float* st2, int final_stage) {
  __shared__ float smat[16][16][16];   // [pt][k][m] normalized+relu
  __shared__ float smp[16][16];        // [pt][k]
  __shared__ float sdw[16][16];        // [pt][m]
  __shared__ float sfc[512];           // fc [16][32]
  __shared__ float sSum[16], sSq[16];
  int t = threadIdx.x, pl = t >> 4, k = t & 15;
  int P = blockIdx.x * 16 + pl;
  int b = P / Ns;
  if (t < 16) { sSum[t] = 0.f; sSq[t] = 0.f; }
  for (int q = t; q < 512; q += 256) sfc[q] = fc[q];
  sdw[pl][k] = dw[(long)P * 16 + k];
  float mu = mean[b * 16 + k], rs = rstd[b * 16 + k];
  float gg = g[k], bv = bb[k];
  for (int m = 0; m < 16; ++m) {
    float y = mat[(long)P * 256 + k * 16 + m];
    y = fmaxf((y - mu) * rs * gg + bv, 0.f);
    smat[pl][k][m] = y;
  }
  __syncthreads();
  float mx = -3.4e38f;
  for (int m = 0; m < 16; ++m) mx = fmaxf(mx, smat[pl][k][m] * sdw[pl][m]);
  smp[pl][k] = mx;
  __syncthreads();
  float lS = 0.f, lQ = 0.f;
  for (int m = 0; m < 16; ++m) {
    float acc = 0.f;
#pragma unroll
    for (int c = 0; c < 16; ++c)
      acc += sfc[k * 32 + c] * smat[pl][c][m] + sfc[k * 32 + 16 + c] * smp[pl][c];
    if (final_stage) acc = fmaxf(acc, 0.f) * sdw[pl][m];
    mat[(long)P * 256 + k * 16 + m] = acc;
    lS += acc; lQ += acc * acc;
  }
  if (!final_stage) {
    atomicAdd(&sSum[k], lS);
    atomicAdd(&sSq[k], lQ);
    __syncthreads();
    int b0 = (blockIdx.x * 16) / Ns;
    if (t < 16) {
      atomicAdd(&st1[b0 * 16 + t], sSum[t]);
      atomicAdd(&st2[b0 * 16 + t], sSq[t]);
    }
  }
}

// ---------------------------------------------------------------------------
// feat[pt, c*16+k] = sum_m x[b,c,idx[pt,m]] * mat[pt,k,m]
// Emitted directly as packed f16 point-major [pt][Kp] (zero-padded).
// ---------------------------------------------------------------------------
__global__ void k_feat(const float* x, const int* idx, const float* mat,
                       int C, int Cp, int Kp, int Np, int Ns, _Float16* feat) {
  __shared__ float smat[256];
  __shared__ int   sidx[16];
  int P = blockIdx.x;
  int b = P / Ns;
  int t = threadIdx.x;
  if (t < 16) sidx[t] = idx[(long)P * 16 + t];
  smat[t] = mat[(long)P * 256 + t];
  __syncthreads();
  int k = t & 15;
  for (int c = t >> 4; c < Cp; c += 16) {
    float acc = 0.f;
    if (c < C) {
      const float* xr = x + ((long)b * C + c) * Np;
#pragma unroll
      for (int m = 0; m < 16; ++m) acc += xr[sidx[m]] * smat[k * 16 + m];
    }
    feat[(long)P * Kp + c * 16 + k] = (_Float16)acc;
  }
}

// ---------------------------------------------------------------------------
// Host orchestration
// ---------------------------------------------------------------------------
static inline int cdiv(int a, int b) { return (a + b - 1) / b; }

extern "C" void kernel_launch(void* const* d_in, const int* in_sizes, int n_in,
                              void* d_out, int out_size, void* d_ws, size_t ws_size,
                              hipStream_t stream) {
  (void)in_sizes; (void)n_in; (void)out_size; (void)ws_size;
  const int N0 = 8192, N1 = 2048, N2 = 512, N3 = 128, N4 = 32;

  auto F = [&](int i) { return (const float*)d_in[i]; };

  // -------- parameter leaf walk (setup_inputs insertion order) --------
  int li = 0;
  const float* x_raw   = F(li++);
  const float* pos_raw = F(li++);
  int cv0_base = li; li += 10;                 // top fka: cv,fc1,fc2,fc3,g1,b1,g2,b2,alpha,beta
  const float* bn0g = F(li++);
  const float* bn0b = F(li++);
  const bool bsc[9] = {false, true, false, true, false, true, false, true, false};
  int blk[9];
  for (int i = 0; i < 9; ++i) { blk[i] = li; li += bsc[i] ? 24 : 20; }
  const float* fcw = F(li++);
  const float* fcb = F(li++);

  // -------- workspace bump allocator --------
  char* wp = (char*)d_ws;
  auto alloc = [&](size_t bytes) -> void* {
    void* p = (void*)wp;
    wp += (bytes + 255) & ~(size_t)255;
    return p;
  };
  float* posact = (float*)alloc((size_t)BATCH * 3 * N0 * 4);
  float* xact   = (float*)alloc((size_t)BATCH * 3 * N0 * 4);
  float* s1 = (float*)alloc((size_t)BATCH * 3 * N1 * 4);
  float* s2 = (float*)alloc((size_t)BATCH * 3 * N2 * 4);
  float* s3 = (float*)alloc((size_t)BATCH * 3 * N3 * 4);
  float* s4 = (float*)alloc((size_t)BATCH * 3 * N4 * 4);
  int*   fpsidx = (int*)alloc((size_t)BATCH * N1 * 4);
  float* mind   = (float*)alloc((size_t)BATCH * N0 * 4);
  int* ids00 = (int*)alloc((size_t)BATCH * N0 * 16 * 4);
  int* ids01 = (int*)alloc((size_t)BATCH * N1 * 16 * 4);
  int* ids11 = (int*)alloc((size_t)BATCH * N1 * 16 * 4);
  int* ids12 = (int*)alloc((size_t)BATCH * N2 * 16 * 4);
  int* ids22 = (int*)alloc((size_t)BATCH * N2 * 16 * 4);
  int* ids23 = (int*)alloc((size_t)BATCH * N3 * 16 * 4);
  int* ids33 = (int*)alloc((size_t)BATCH * N3 * 16 * 4);
  int* ids34 = (int*)alloc((size_t)BATCH * N4 * 16 * 4);
  int* ids44 = (int*)alloc((size_t)BATCH * N4 * 16 * 4);
  float* st    = (float*)alloc(64 * 4);          // st1 | st2
  float* meanb = (float*)alloc(32 * 4);
  float* rstdb = (float*)alloc(32 * 4);
  float* dwb   = (float*)alloc((size_t)BATCH * N0 * 16 * 4);
  float* matb  = (float*)alloc((size_t)BATCH * N0 * 256 * 4);          // 16 MB
  _Float16* Xh = (_Float16*)alloc((size_t)BATCH * N0 * 512 * 2);       // 16 MB packed B
  _Float16* Wh = (_Float16*)alloc((size_t)1024 * 8192 * 2);            // 16 MB packed A
  size_t ACT = (size_t)BATCH * 1024 * 1024 * 4;                        // 8 MB
  float* XA = (float*)alloc(ACT);
  float* XB = (float*)alloc(ACT);
  float* H1 = (float*)alloc(ACT);
  float* H2 = (float*)alloc(ACT / 2);
  float* H3 = (float*)alloc(ACT / 2);
  float* H4 = (float*)alloc(ACT / 2);
  float* xm = (float*)alloc((size_t)BATCH * 1024 * 4);

  // -------- helpers --------
  // GEMM with packed weights; Xh must already hold packed f16 [Nc][Kp].
  auto gemm_packed = [&](const float* Wp, float* Yp,
                         const float* bias, const float* g, const float* bb,
                         int Mr, int Kd, int Kp, int Nper, int relu) {
    int Nc = BATCH * Nper;
    k_pack_w<<<cdiv(Mr * Kp, 256), 256, 0, stream>>>(Wp, Wh, Kd, Kp, Mr * Kp);
    dim3 grid(Nc / 64, Mr / 32);
    k_gemm_wmma<<<grid, 128, 0, stream>>>(Wh, Xh, Yp, bias, g, bb,
                                          Mr, Kp, Nc, Nper, relu);
  };
  // full GEMM from f32 activation-layout input
  auto gemm = [&](const float* Wp, const float* Xp, float* Yp,
                  const float* bias, const float* g, const float* bb,
                  int Mr, int Kd, int Nper, int relu) {
    int Kp = (Kd + 31) & ~31;
    int Nc = BATCH * Nper;
    k_pack_x<<<cdiv(Nc * Kp, 256), 256, 0, stream>>>(Xp, Xh, Kd, Kp, Nper, Nc * Kp);
    gemm_packed(Wp, Yp, bias, g, bb, Mr, Kd, Kp, Nper, relu);
  };

  auto fka = [&](int pb, const float* xin, int C, int Cout,
                 const float* possrc, const float* sup, const int* idx,
                 int Np, int Ns, const float* bg, const float* bbp, float* yout) {
    const float* cv  = F(pb + 0);
    const float* fc1 = F(pb + 1);
    const float* fc2 = F(pb + 2);
    const float* fc3 = F(pb + 3);
    const float* g1  = F(pb + 4);
    const float* b1  = F(pb + 5);
    const float* g2  = F(pb + 6);
    const float* b2  = F(pb + 7);
    const float* al  = F(pb + 8);
    const float* be  = F(pb + 9);
    int NP = BATCH * Ns;
    int Kd = C * 16;
    int Kp = (Kd + 31) & ~31;
    int Cp = Kp / 16;
    k_zero<<<1, 256, 0, stream>>>(st, 64);
    k_fka_a<<<NP / 16, 256, 0, stream>>>(possrc, sup, idx, fc1, al, be,
                                         Np, Ns, dwb, matb, st, st + 32);
    k_stats<<<1, 32, 0, stream>>>(st, st + 32, Ns * 16, meanb, rstdb);
    k_fka_bc<<<NP / 16, 256, 0, stream>>>(meanb, rstdb, g1, b1, fc2, dwb, matb,
                                          Ns, st, st + 32, 0);
    k_stats<<<1, 32, 0, stream>>>(st, st + 32, Ns * 16, meanb, rstdb);
    k_fka_bc<<<NP / 16, 256, 0, stream>>>(meanb, rstdb, g2, b2, fc3, dwb, matb,
                                          Ns, nullptr, nullptr, 1);
    k_feat<<<NP, 256, 0, stream>>>(xin, idx, matb, C, Cp, Kp, Np, Ns, Xh);
    gemm_packed(cv, yout, nullptr, bg, bbp, Cout, Kd, Kp, Ns, 1);
  };

  // -------- transposes --------
  k_transpose<<<cdiv(BATCH * N0 * 3, 256), 256, 0, stream>>>(x_raw, xact, N0, 3);
  k_transpose<<<cdiv(BATCH * N0 * 3, 256), 256, 0, stream>>>(pos_raw, posact, N0, 3);

  // -------- FPS hierarchy --------
  k_fps<<<BATCH, 256, 0, stream>>>(posact, N0, N1, fpsidx, mind);
  k_gather3<<<cdiv(BATCH * 3 * N1, 256), 256, 0, stream>>>(posact, fpsidx, N0, N1, s1);
  k_fps<<<BATCH, 256, 0, stream>>>(s1, N1, N2, fpsidx, mind);
  k_gather3<<<cdiv(BATCH * 3 * N2, 256), 256, 0, stream>>>(s1, fpsidx, N1, N2, s2);
  k_fps<<<BATCH, 256, 0, stream>>>(s2, N2, N3, fpsidx, mind);
  k_gather3<<<cdiv(BATCH * 3 * N3, 256), 256, 0, stream>>>(s2, fpsidx, N2, N3, s3);
  k_fps<<<BATCH, 256, 0, stream>>>(s3, N3, N4, fpsidx, mind);
  k_gather3<<<cdiv(BATCH * 3 * N4, 256), 256, 0, stream>>>(s3, fpsidx, N3, N4, s4);

  // -------- KNN graphs --------
  k_knn<<<cdiv(BATCH * N0, 256), 256, 0, stream>>>(posact, posact, N0, N0, ids00);
  k_knn<<<cdiv(BATCH * N1, 256), 256, 0, stream>>>(posact, s1, N0, N1, ids01);
  k_knn<<<cdiv(BATCH * N1, 256), 256, 0, stream>>>(s1, s1, N1, N1, ids11);
  k_knn<<<cdiv(BATCH * N2, 256), 256, 0, stream>>>(s1, s2, N1, N2, ids12);
  k_knn<<<cdiv(BATCH * N2, 256), 256, 0, stream>>>(s2, s2, N2, N2, ids22);
  k_knn<<<cdiv(BATCH * N3, 256), 256, 0, stream>>>(s2, s3, N2, N3, ids23);
  k_knn<<<cdiv(BATCH * N3, 256), 256, 0, stream>>>(s3, s3, N3, N3, ids33);
  k_knn<<<cdiv(BATCH * N4, 256), 256, 0, stream>>>(s3, s4, N3, N4, ids34);
  k_knn<<<cdiv(BATCH * N4, 256), 256, 0, stream>>>(s4, s4, N4, N4, ids44);

  // -------- stem: relu(bn0(fka_cv0(x))) --------
  fka(cv0_base, xact, 3, 64, posact, posact, ids00, N0, N0, bn0g, bn0b, XA);

  // -------- residual blocks --------
  struct Cfg { int cin, cout, np, ns; const float *pos, *sup; const int* idx; };
  Cfg cfg[9] = {
      {64,   64,   N0, N0, posact, posact, ids00},
      {64,   128,  N0, N1, posact, s1,     ids01},
      {128,  128,  N1, N1, s1,     s1,     ids11},
      {128,  256,  N1, N2, s1,     s2,     ids12},
      {256,  256,  N2, N2, s2,     s2,     ids22},
      {256,  512,  N2, N3, s2,     s3,     ids23},
      {512,  512,  N3, N3, s3,     s3,     ids33},
      {512,  1024, N3, N4, s3,     s4,     ids34},
      {1024, 1024, N4, N4, s4,     s4,     ids44}};

  float* cur = XA;
  float* oth = XB;
  for (int i = 0; i < 9; ++i) {
    int base = blk[i];
    const float* cv0w  = F(base + 0);
    const float* cv0bv = F(base + 1);
    const float* b0g = F(base + 2);
    const float* b0b = F(base + 3);
    int fkabase = base + 4;
    const float* b1g = F(base + 14);
    const float* b1b = F(base + 15);
    const float* cv2w  = F(base + 16);
    const float* cv2bv = F(base + 17);
    const float* b2g = F(base + 18);
    const float* b2b = F(base + 19);
    Cfg c = cfg[i];
    int Ch = c.cin / 2;

    // main path
    gemm(cv0w, cur, H1, cv0bv, b0g, b0b, Ch, c.cin, c.np, 1);
    fka(fkabase, H1, Ch, Ch, c.pos, c.sup, c.idx, c.np, c.ns, b1g, b1b, H2);
    gemm(cv2w, H2, H3, cv2bv, b2g, b2b, c.cout, Ch, c.ns, 0);

    // shortcut
    const float* xs = cur;
    if (bsc[i]) {
      const float* scw = F(base + 20);
      const float* scb = F(base + 21);
      const float* bsg = F(base + 22);
      const float* bsb = F(base + 23);
      gemm(scw, cur, H1, scb, bsg, bsb, c.cout, c.cin, c.np, 0);
      xs = H1;
    }
    if (c.np != c.ns) {
      k_maxpool<<<cdiv(BATCH * c.cout * c.ns, 256), 256, 0, stream>>>(
          (const float*)xs, c.idx, c.cout, c.np, c.ns, H4);
      xs = H4;
    }
    k_add_relu<<<cdiv(BATCH * c.cout * c.ns, 256), 256, 0, stream>>>(
        H3, xs, oth, BATCH * c.cout * c.ns);
    float* tmp = cur; cur = oth; oth = tmp;
  }

  // -------- head: mean over points + FC --------
  k_mean<<<cdiv(BATCH * 1024, 256), 256, 0, stream>>>(cur, 1024, N4, xm);
  k_fc<<<cdiv(BATCH * 40, 64), 64, 0, stream>>>(xm, fcw, fcb, (float*)d_out, 1024, 40);
}
